// DenseGGNN_50337016709455
// MI455X (gfx1250) — matrix operand order repaired and checked by reading, not compiled, not run on hardware
//
#include <hip/hip_runtime.h>
#include <hip/hip_bf16.h>
#include <stdint.h>

// ---------------------------------------------------------------------------
// Types
// ---------------------------------------------------------------------------
typedef __bf16 bf16;
typedef __bf16 v8bf  __attribute__((ext_vector_type(8)));
typedef __bf16 v16bf __attribute__((ext_vector_type(16)));
typedef float  v8f   __attribute__((ext_vector_type(8)));

#define Bq 32
#define Nq 1024
#define Cq 128
#define Lq 4

// ---------------------------------------------------------------------------
// Helpers
// ---------------------------------------------------------------------------
static __device__ __forceinline__ bf16 f2bf(float f) {
  return (bf16)f;   // native cast -> v_cvt_pk_bf16_f32 pairs (RNE)
}

static __device__ __forceinline__ v8f vzero8() {
  v8f v;
#pragma unroll
  for (int i = 0; i < 8; ++i) v[i] = 0.0f;
  return v;
}

static __device__ __forceinline__ v8f wmma_bf16(v16bf a, v16bf b, v8f c) {
  // D = A(16x32 bf16) * B(32x16 bf16) + C(16x16 f32)
  return __builtin_amdgcn_wmma_f32_16x16x32_bf16(
      /*neg_a=*/false, a, /*neg_b=*/false, b,
      /*c_mod=*/(short)0, c, /*reuse_a=*/false, /*reuse_b=*/false);
}

// A-matrix 16x32 bf16: lane L -> row L&15;
// lanes 0-15 hold K = k0+[0..15]; lanes 16-31: k0+[16..31].
static __device__ __forceinline__ v16bf load_A(const bf16* A, int lda,
                                               int lane, int k0) {
  int row = lane & 15;
  int kb  = (lane >> 4) * 16;
  const bf16* p = A + (size_t)row * lda + k0 + kb;
  v8bf lo = *(const v8bf*)p;
  v8bf hi = *(const v8bf*)(p + 8);
  v16bf r;
#pragma unroll
  for (int i = 0; i < 8; ++i) { r[i] = lo[i]; r[i + 8] = hi[i]; }
  return r;
}

// B-matrix 32x16 bf16: lane holds one column's K-run.  We store B transposed
// (column n == contiguous row n of Bt), so the load is one 32-byte vector.
static __device__ __forceinline__ v16bf load_BT(const bf16* Bt, int ldb,
                                                int lane, int k0) {
  int col = lane & 15;
  int kb  = (lane >> 4) * 16;
  return *(const v16bf*)(Bt + (size_t)col * ldb + k0 + kb);
}

static __device__ __forceinline__ float sigmoidf_(float x) {
  return 1.0f / (1.0f + __expf(-x));
}
static __device__ __forceinline__ float tanhf_(float x) {
  return 1.0f - 2.0f / (__expf(2.0f * x) + 1.0f);
}

// CDNA5 async Global->LDS copy (no VGPR staging; tracked by ASYNCcnt).
// GVS form: dst = LDS byte offset (VGPR), voff = per-lane byte offset (VGPR),
// base = uniform 64-bit global base (SGPR pair).
static __device__ __forceinline__ void async_ld_b128(unsigned lds_off,
                                                     unsigned gbl_off,
                                                     const void* base) {
  asm volatile("global_load_async_to_lds_b128 %0, %1, %2 offset:0"
               :: "v"(lds_off), "v"(gbl_off), "s"(base)
               : "memory");
}

// ---------------------------------------------------------------------------
// Prep kernels
// ---------------------------------------------------------------------------
__global__ void prep_x_kernel(const float* __restrict__ x,
                              float* __restrict__ hf,
                              bf16* __restrict__ hbf, int n) {
  int i = blockIdx.x * blockDim.x + threadIdx.x;
  if (i < n) { float v = x[i]; hf[i] = v; hbf[i] = f2bf(v); }
}

__global__ void cvt_bf16_kernel(const float* __restrict__ src,
                                bf16* __restrict__ dst, int n) {
  int i = blockIdx.x * blockDim.x + threadIdx.x;
  if (i < n) dst[i] = f2bf(src[i]);
}

// weight [L][C][C] -> Wt [L][C][C] with Wt[l][d][c] = W[l][c][d] (bf16)
__global__ void prep_wt_kernel(const float* __restrict__ w,
                               bf16* __restrict__ wt, int n) {
  int i = blockIdx.x * blockDim.x + threadIdx.x;
  if (i >= n) return;
  int l = i / (Cq * Cq);
  int r = i - l * (Cq * Cq);
  int c = r / Cq;
  int d = r - c * Cq;
  wt[(size_t)l * Cq * Cq + (size_t)d * Cq + c] = f2bf(w[i]);
}

// adjT[b][n][j] = bf16(adj[b][j][n]) — 32x32 LDS tile transpose
__global__ void prep_adj_kernel(const float* __restrict__ adj,
                                bf16* __restrict__ adjT) {
  __shared__ float t[32][33];
  int b  = blockIdx.z;
  int n0 = blockIdx.x * 32;
  int j0 = blockIdx.y * 32;
  int tx = threadIdx.x, ty = threadIdx.y;   // (32, 8)
  const float* src = adj + ((size_t)b << 20);
#pragma unroll
  for (int i = 0; i < 4; ++i) {
    int j = ty + 8 * i;
    t[j][tx] = src[(size_t)(j0 + j) * Nq + n0 + tx];
  }
  __syncthreads();
  bf16* dst = adjT + ((size_t)b << 20);
#pragma unroll
  for (int i = 0; i < 4; ++i) {
    int n = ty + 8 * i;
    dst[(size_t)(n0 + n) * Nq + j0 + tx] = f2bf(t[tx][n]);
  }
}

// ---------------------------------------------------------------------------
// hw_t[b][d][n] = sum_c h[b][n][c] * W[c][d]   (output stored transposed)
// block = 8 waves; all waves share one 16-row tile, each owns one d-tile.
// ---------------------------------------------------------------------------
__global__ void gemm_hw_kernel(const bf16* __restrict__ hbf,
                               const bf16* __restrict__ wt,   // [C][C] Wt
                               bf16* __restrict__ hwt) {
  int wid  = threadIdx.x >> 5;
  int lane = threadIdx.x & 31;
  int rt = blockIdx.x * 16;      // global row tile (b*N + n)
  int ct = wid * 16;             // d tile
  const bf16* A  = hbf + (size_t)rt * Cq;
  const bf16* Bt = wt + (size_t)ct * Cq;
  v8f acc = vzero8();
#pragma unroll
  for (int k0 = 0; k0 < Cq; k0 += 32)
    acc = wmma_bf16(load_A(A, Cq, lane, k0), load_BT(Bt, Cq, lane, k0), acc);
  // D layout: lane -> col = lane&15, rows = 8*(lane>>4)+v  -> transposed pack
  int b = rt >> 10;
  int n = rt & (Nq - 1);
  bf16* out = hwt + ((size_t)b * Cq + ct + (lane & 15)) * Nq
                  + n + (lane >> 4) * 8;
  v8bf pk;
#pragma unroll
  for (int v = 0; v < 8; ++v) pk[v] = f2bf(acc[v]);
  *(v8bf*)out = pk;
}

// ---------------------------------------------------------------------------
// Fused message + GRU kernel for one (b, 16-node) tile.
//
// Phase 1 (message):  m[n][d] = sum_j adjT[b][n][j] * hw_t[b][d][j], K=1024.
//   The 16x1024 adjacency slab (32 KB) is staged in LDS via the CDNA5 async
//   Global->LDS path in two 16 KB halves, software-pipelined against the
//   WMMA loop; 8 waves reuse the slab (one d-tile each).  The m tile never
//   touches HBM: accumulators are converted to bf16 straight into LDS.
// Phase 2 (GRU):  wave w owns c-tile w; 6 gate tiles (24 WMMAs) from the
//   LDS m/h tiles, then the sigmoid/tanh blend in registers.  Updates h
//   (f32 + bf16) in place; optionally writes the masked output.
// ---------------------------------------------------------------------------
__global__ void msg_gru_kernel(const bf16* __restrict__ adjT,
                               const bf16* __restrict__ hwt,
                               bf16* __restrict__ hbf,       // read + write
                               float* __restrict__ hf,       // read + write
                               const bf16* __restrict__ wih, // [3C][C] bf16
                               const bf16* __restrict__ whh, // [3C][C] bf16
                               const float* __restrict__ bih,
                               const float* __restrict__ bhh,
                               const float* __restrict__ mask,
                               float* __restrict__ out) {    // null except last
  __shared__ bf16 As[16 * Nq];                       // 32 KB adjacency slab
  __shared__ bf16 Ms[16 * Cq];                       // 4 KB message tile
  __shared__ bf16 Hs[16 * Cq];                       // 4 KB h tile
  int b  = blockIdx.x >> 6;
  int nt = (blockIdx.x & 63) * 16;
  size_t rowbase = (size_t)b * Nq + nt;              // global row = b*N + n

  const bf16* src = adjT + ((size_t)b << 20) + (size_t)nt * Nq;
  unsigned lds_base = (unsigned)(uintptr_t)(void*)As;  // addr[31:0] = LDS off
  int t = threadIdx.x;
  // 2048 x 16B chunks; 8 per thread; j=0..3 -> first 16 KB, j=4..7 -> second.
#pragma unroll
  for (int j = 0; j < 8; ++j) {
    unsigned off = (unsigned)(t + j * 256) * 16u;
    async_ld_b128(lds_base + off, off, src);
  }
  // stage the h tile while the async copy flies
  *(v8bf*)(Hs + (size_t)t * 8) =
      *(const v8bf*)(hbf + rowbase * Cq + (size_t)t * 8);

  int wid  = threadIdx.x >> 5;
  int lane = threadIdx.x & 31;
  int ct = wid * 16;                                 // d tile (phase1) = c tile (phase2)
  const bf16* Bt = hwt + ((size_t)b * Cq + ct) * Nq;
  v8f a0 = vzero8(), a1 = vzero8();

#pragma unroll
  for (int half = 0; half < 2; ++half) {
    if (half == 0)
      asm volatile("s_wait_asynccnt 0x4" ::: "memory");  // first 16 KB landed
    else
      asm volatile("s_wait_asynccnt 0x0" ::: "memory");  // rest landed
    __syncthreads();
    int kbeg = half * 512;
#pragma unroll 4
    for (int k0 = kbeg; k0 < kbeg + 512; k0 += 64) {
      __builtin_prefetch(Bt + (size_t)(lane & 15) * Nq + k0 + 256, 0, 0);
      a0 = wmma_bf16(load_A(As, Nq, lane, k0),
                     load_BT(Bt, Nq, lane, k0), a0);
      a1 = wmma_bf16(load_A(As, Nq, lane, k0 + 32),
                     load_BT(Bt, Nq, lane, k0 + 32), a1);
    }
  }
  v8f acc = a0 + a1;
  // m tile straight to LDS: lane -> col = ct+(lane&15), rows = 8*(lane>>4)+v
  {
    bf16* mp = Ms + (size_t)((lane >> 4) * 8) * Cq + ct + (lane & 15);
#pragma unroll
    for (int v = 0; v < 8; ++v) mp[(size_t)v * Cq] = f2bf(acc[v]);
  }
  __syncthreads();

  // ---- Phase 2: GRU ----
  v8f xr = vzero8(), xz = vzero8(), xn = vzero8();
  v8f hr = vzero8(), hz = vzero8(), hn = vzero8();
#pragma unroll
  for (int k0 = 0; k0 < Cq; k0 += 32) {
    v16bf am = load_A(Ms, Cq, lane, k0);
    v16bf ah = load_A(Hs, Cq, lane, k0);
    xr = wmma_bf16(am, load_BT(wih + (size_t)(0 * Cq + ct) * Cq, Cq, lane, k0), xr);
    xz = wmma_bf16(am, load_BT(wih + (size_t)(1 * Cq + ct) * Cq, Cq, lane, k0), xz);
    xn = wmma_bf16(am, load_BT(wih + (size_t)(2 * Cq + ct) * Cq, Cq, lane, k0), xn);
    hr = wmma_bf16(ah, load_BT(whh + (size_t)(0 * Cq + ct) * Cq, Cq, lane, k0), hr);
    hz = wmma_bf16(ah, load_BT(whh + (size_t)(1 * Cq + ct) * Cq, Cq, lane, k0), hz);
    hn = wmma_bf16(ah, load_BT(whh + (size_t)(2 * Cq + ct) * Cq, Cq, lane, k0), hn);
  }

  int col  = ct + (lane & 15);
  float bri = bih[0 * Cq + col], bzi = bih[1 * Cq + col], bni = bih[2 * Cq + col];
  float brh = bhh[0 * Cq + col], bzh = bhh[1 * Cq + col], bnh = bhh[2 * Cq + col];
  size_t r0 = rowbase + (lane >> 4) * 8;
#pragma unroll
  for (int v = 0; v < 8; ++v) {
    size_t idx = (r0 + v) * Cq + col;
    float hold = hf[idx];
    float r = sigmoidf_(xr[v] + bri + hr[v] + brh);
    float z = sigmoidf_(xz[v] + bzi + hz[v] + bzh);
    float n = tanhf_(xn[v] + bni + r * (hn[v] + bnh));
    float hnew = (1.0f - z) * n + z * hold;
    hf[idx]  = hnew;
    hbf[idx] = f2bf(hnew);
    if (out) out[idx] = hnew * mask[r0 + v];
  }
}

// ---------------------------------------------------------------------------
// Host launcher
// ---------------------------------------------------------------------------
extern "C" void kernel_launch(void* const* d_in, const int* in_sizes, int n_in,
                              void* d_out, int out_size, void* d_ws, size_t ws_size,
                              hipStream_t stream) {
  const float* x    = (const float*)d_in[0];
  const float* adj  = (const float*)d_in[1];
  const float* mask = (const float*)d_in[2];
  const float* w    = (const float*)d_in[3];
  const float* wih  = (const float*)d_in[4];
  const float* whh  = (const float*)d_in[5];
  const float* bih  = (const float*)d_in[6];
  const float* bhh  = (const float*)d_in[7];
  float* out = (float*)d_out;

  // workspace layout (256B aligned)
  char* ws = (char*)d_ws;
  size_t off = 0;
  auto alloc = [&](size_t bytes) {
    char* p = ws + off;
    off += (bytes + 255) & ~size_t(255);
    return p;
  };
  bf16*  adjT   = (bf16*)alloc((size_t)Bq * Nq * Nq * 2);       // 67 MB
  float* h_f32  = (float*)alloc((size_t)Bq * Nq * Cq * 4);      // 16.8 MB
  bf16*  h_bf   = (bf16*)alloc((size_t)Bq * Nq * Cq * 2);       // 8.4 MB
  bf16*  hwt    = (bf16*)alloc((size_t)Bq * Cq * Nq * 2);       // 8.4 MB
  bf16*  wt_bf  = (bf16*)alloc((size_t)Lq * Cq * Cq * 2);
  bf16*  wih_bf = (bf16*)alloc((size_t)Lq * 3 * Cq * Cq * 2);
  bf16*  whh_bf = (bf16*)alloc((size_t)Lq * 3 * Cq * Cq * 2);
  (void)ws_size; (void)n_in; (void)in_sizes; (void)out_size;

  // one-time prep
  prep_adj_kernel<<<dim3(Nq / 32, Nq / 32, Bq), dim3(32, 8), 0, stream>>>(adj, adjT);
  {
    int n = Bq * Nq * Cq;
    prep_x_kernel<<<(n + 255) / 256, 256, 0, stream>>>(x, h_f32, h_bf, n);
  }
  {
    int n = Lq * Cq * Cq;
    prep_wt_kernel<<<(n + 255) / 256, 256, 0, stream>>>(w, wt_bf, n);
  }
  {
    int n = Lq * 3 * Cq * Cq;
    cvt_bf16_kernel<<<(n + 255) / 256, 256, 0, stream>>>(wih, wih_bf, n);
    cvt_bf16_kernel<<<(n + 255) / 256, 256, 0, stream>>>(whh, whh_bf, n);
  }

  const int rowTiles = Bq * Nq / 16;   // 2048
  for (int l = 0; l < Lq; ++l) {
    gemm_hw_kernel<<<rowTiles, 256, 0, stream>>>(
        h_bf, wt_bf + (size_t)l * Cq * Cq, hwt);
    msg_gru_kernel<<<rowTiles, 256, 0, stream>>>(
        adjT, hwt, h_bf, h_f32,
        wih_bf + (size_t)l * 3 * Cq * Cq, whh_bf + (size_t)l * 3 * Cq * Cq,
        bih + (size_t)l * 3 * Cq, bhh + (size_t)l * 3 * Cq,
        mask, (l == Lq - 1) ? out : nullptr);
  }
}
